// MDTA_84009560310422
// MI455X (gfx1250) — matrix operand order repaired and checked by reading, not compiled, not run on hardware
//
#include <hip/hip_runtime.h>
#include <hip/hip_bf16.h>

typedef __attribute__((ext_vector_type(16))) _Float16 v16h;
typedef __attribute__((ext_vector_type(8)))  float    v8f;

#define DIM   192
#define HEADS 8
#define HD    24
#define HDP   32          // head_dim padded to WMMA K depth
#define NPIX  4096
#define ATT_SCALE 0.20412414523193154f   // 24^-0.5

union frag_u  { v16h v; uint4 q[2]; };
union facc_u  { v8f  v; uint4 q[2]; };

// ---- CDNA5 async global->LDS copy (ASYNCcnt-tracked) ----
__device__ __forceinline__ void async_b128_to_lds(const _Float16* gptr, void* lds_ptr) {
  const unsigned lds_off = (unsigned)(size_t)lds_ptr;   // LDS aperture: offset in addr[31:0]
  asm volatile("global_load_async_to_lds_b128 %0, %1, off"
               :: "v"(lds_off), "v"(gptr) : "memory");
}
__device__ __forceinline__ void wait_async_lds() {
  asm volatile("s_wait_asynccnt 0" ::: "memory");
}

// ---------------- WMMA A fragment (16x32 f16) from global, row-major ----------------
// lane m = lane&15; k = kbase + (h<8 ? h : h+8), kbase = (lane>=16)?8:0
__device__ __forceinline__ v16h load_a_frag_g(const _Float16* __restrict__ A, int lda,
                                              int row0, int k0) {
  const int lane = threadIdx.x & 31;
  const _Float16* p = A + (size_t)(row0 + (lane & 15)) * lda + k0 + ((lane & 16) >> 1);
  frag_u u;
  u.q[0] = *(const uint4*)p;          // k = kb .. kb+7
  u.q[1] = *(const uint4*)(p + 16);   // k = kb+16 .. kb+23
  return u.v;
}

// ---------------- kernel 0: fp32 -> f16 conversions (x goes pixel-major) -------------
__global__ void convert_inputs(const float* __restrict__ x, const float* __restrict__ wq,
                               const float* __restrict__ wp, _Float16* __restrict__ xt,
                               _Float16* __restrict__ wqh, _Float16* __restrict__ wph) {
  const int i = blockIdx.x * 256 + threadIdx.x;
  if (i < DIM * NPIX) {                      // xt[p][c] = x[c][p]
    const int p = i / DIM, c = i - p * DIM;
    xt[i] = (_Float16)x[(size_t)c * NPIX + p];
  }
  if (i < 3 * DIM * DIM)  wqh[i] = (_Float16)wq[i];
  if (i < DIM * DIM)      wph[i] = (_Float16)wp[i];
}

// ------- WMMA GEMM (transposed form): C[o][p] f32, computed as D[p][o] tiles --------
// Xt: [NPIX][192] pixel-major f16 activations; W: [O][192] f16 weights.
// grid.x = pixel tile (NPIX/16); wave handles outchan tile = blockIdx.y*4 + wave.
// 16x192 pixel panel async-staged in LDS, shared by the 4 waves; all accesses b128.
__global__ __launch_bounds__(128) void wmma_gemm_xt(const _Float16* __restrict__ Xt,
                                                    const _Float16* __restrict__ W,
                                                    float* __restrict__ C) {
  __shared__ __align__(16) _Float16 Xs[16][DIM];   // 6 KB pixel panel
  const int tid  = threadIdx.x;
  const int pt   = blockIdx.x;
  const int ot   = blockIdx.y * 4 + (tid >> 5);
  const int lane = tid & 31;

  // async-stage Xt[pt*16 .. +16)[0..192): 384 16-byte chunks over 128 threads
#pragma unroll
  for (int j = 0; j < 3; ++j) {
    const int c = tid + 128 * j;
    const int r = c / 24, o8 = (c % 24) * 8;     // 24 chunks per 192-half row
    async_b128_to_lds(Xt + (size_t)(pt * 16 + r) * DIM + o8, &Xs[r][o8]);
  }
  wait_async_lds();
  __syncthreads();

  v8f acc = {};
#pragma unroll
  for (int ks = 0; ks < 6; ++ks) {
    // A-frag (pixels x k) from LDS: contiguous per lane
    frag_u a;
    const _Float16* ap = &Xs[lane & 15][ks * 32 + ((lane & 16) >> 1)];
    a.q[0] = *(const uint4*)ap;
    a.q[1] = *(const uint4*)(ap + 16);
    // B-frag: B[k][n] = W[ot*16+n][k] -> contiguous run of a weight row
    frag_u b;
    const _Float16* bp = W + (size_t)(ot * 16 + (lane & 15)) * DIM + ks * 32 + ((lane & 16) >> 1);
    b.q[0] = *(const uint4*)bp;
    b.q[1] = *(const uint4*)(bp + 16);
    acc = __builtin_amdgcn_wmma_f32_16x16x32_f16(false, a.v, false, b.v, (short)0, acc,
                                                 false, false);
  }
  // D rows = pixels, cols = outchans; lane owns one o and 8 consecutive pixels
  const int o    = ot * 16 + (lane & 15);
  const int pix0 = pt * 16 + ((lane & 16) >> 1);
  facc_u st; st.v = acc;
  uint4* cp = (uint4*)(C + (size_t)o * NPIX + pix0);
  cp[0] = st.q[0];
  cp[1] = st.q[1];
}

// ---------------- kernel 2: depthwise 3x3, SAME padding, fp32 ----------------
__global__ void dwconv3x3(const float* __restrict__ in, const float* __restrict__ w,
                          float* __restrict__ out) {
  const int i = blockIdx.x * 256 + threadIdx.x;   // c*4096 + p
  const int c = i >> 12, p = i & 4095;
  const int y = p >> 6, x = p & 63;
  const float* wc = w + c * 9;
  const float* ic = in + (size_t)c * NPIX;
  float s = 0.f;
#pragma unroll
  for (int dy = -1; dy <= 1; ++dy) {
    const int yy = y + dy;
    if ((unsigned)yy >= 64u) continue;
#pragma unroll
    for (int dx = -1; dx <= 1; ++dx) {
      const int xx = x + dx;
      if ((unsigned)xx >= 64u) continue;
      s += wc[(dy + 1) * 3 + (dx + 1)] * ic[yy * 64 + xx];
    }
  }
  out[i] = s;
}

// ------- kernel 3: L2-normalize q,k over head_dim; pack Q/K/V f16 [8][4096][32] ------
__global__ void normalize_pack(const float* __restrict__ dw, _Float16* __restrict__ Q,
                               _Float16* __restrict__ K, _Float16* __restrict__ V) {
  const int i = blockIdx.x * 128 + threadIdx.x;   // head*4096 + pixel
  const int h = i >> 12, p = i & 4095;
  const float* qs = dw + (size_t)(h * HD) * NPIX + p;
  const float* ks = dw + (size_t)(DIM + h * HD) * NPIX + p;
  const float* vs = dw + (size_t)(2 * DIM + h * HD) * NPIX + p;
  float qv[HD], kv[HD], qn = 0.f, kn = 0.f;
#pragma unroll
  for (int d = 0; d < HD; ++d) {
    qv[d] = qs[(size_t)d * NPIX]; qn += qv[d] * qv[d];
    kv[d] = ks[(size_t)d * NPIX]; kn += kv[d] * kv[d];
  }
  const float qi = 1.f / fmaxf(sqrtf(qn), 1e-12f);
  const float ki = 1.f / fmaxf(sqrtf(kn), 1e-12f);
  _Float16* qd = Q + (size_t)i * HDP;
  _Float16* kd = K + (size_t)i * HDP;
  _Float16* vd = V + (size_t)i * HDP;
#pragma unroll
  for (int d = 0; d < HDP; ++d) {
    qd[d] = (_Float16)(d < HD ? qv[d] * qi : 0.f);
    kd[d] = (_Float16)(d < HD ? kv[d] * ki : 0.f);
    vd[d] = (_Float16)(d < HD ? vs[(size_t)d * NPIX] : 0.f);
  }
}

// ---------------- kernel 4: flash attention, one head x 64-row block per WG ----------
// Output goes pixel-major: AOt[pixel][192] so the proj GEMM can consume it directly.
__global__ __launch_bounds__(128) void flash_attn(const _Float16* __restrict__ Qg,
                                                  const _Float16* __restrict__ Kg,
                                                  const _Float16* __restrict__ Vg,
                                                  _Float16* __restrict__ AOt) {
  __shared__ __align__(16) _Float16 Ks[32][HDP];      // 32 key rows x 32 d
  __shared__ __align__(16) _Float16 Vt[HDP][32];      // transposed: d x key row
  __shared__ __align__(16) _Float16 Ps[4][16][32];    // per-wave P staging
  const int head = blockIdx.y;
  const int wave = threadIdx.x >> 5;
  const int lane = threadIdx.x & 31;
  const int tid  = threadIdx.x;
  const _Float16* Qh = Qg + (size_t)head * NPIX * HDP;
  const _Float16* Kh = Kg + (size_t)head * NPIX * HDP;
  const _Float16* Vh = Vg + (size_t)head * NPIX * HDP;
  const int row0 = blockIdx.x * 64 + wave * 16;

  const v16h qfrag = load_a_frag_g(Qh, HDP, row0, 0);   // Q tile lives in registers

  v8f o0 = {}, o1 = {};
  float mrun[8], lsum[8];
#pragma unroll
  for (int r = 0; r < 8; ++r) { mrun[r] = -1e30f; lsum[r] = 0.f; }

  for (int jb = 0; jb < NPIX / 32; ++jb) {
    const int j0 = jb * 32;
    __syncthreads();
    { // K tile: 32x32 halfs via CDNA5 async copy (16 B / thread)
      const int r = tid >> 2, c8 = (tid & 3) * 8;
      async_b128_to_lds(Kh + (size_t)(j0 + r) * HDP + c8, &Ks[r][c8]);
    }
    { // V tile, transposed into LDS (elementwise, LDS-bound anyway)
      const int d = tid & 31, rg = (tid >> 5) * 8;
#pragma unroll
      for (int rr = 0; rr < 8; ++rr)
        Vt[d][rg + rr] = Vh[(size_t)(j0 + rg + rr) * HDP + d];
    }
    if (jb + 1 < NPIX / 32)                        // pull next K tile toward L2
      __builtin_prefetch(Kh + (size_t)(j0 + 32) * HDP + tid * 8, 0, 0);
    wait_async_lds();
    __syncthreads();

    // ---- S = (Q Kt) * scale : two 16x16 tiles over this 32-key block ----
    v8f s0 = {}, s1 = {};
    {
      const int n = lane & 15, kb = (lane & 16) >> 1;     // B[k=d][n=key] = K[key][d]
      frag_u b0, b1;
      const _Float16* p0r = &Ks[n][kb];
      const _Float16* p1r = &Ks[16 + n][kb];
      b0.q[0] = *(const uint4*)p0r;  b0.q[1] = *(const uint4*)(p0r + 16);
      b1.q[0] = *(const uint4*)p1r;  b1.q[1] = *(const uint4*)(p1r + 16);
      s0 = __builtin_amdgcn_wmma_f32_16x16x32_f16(false, qfrag, false, b0.v, (short)0,
                                                  s0, false, false);
      s1 = __builtin_amdgcn_wmma_f32_16x16x32_f16(false, qfrag, false, b1.v, (short)0,
                                                  s1, false, false);
    }

    // ---- online softmax (row r of C-tile = VGPR r + 8*lane_hi) ----
    float p0[8], p1[8];
#pragma unroll
    for (int r = 0; r < 8; ++r) {
      const float a = s0[r] * ATT_SCALE, b = s1[r] * ATT_SCALE;
      float tm = fmaxf(a, b);
      tm = fmaxf(tm, __shfl_xor(tm, 1, 16));
      tm = fmaxf(tm, __shfl_xor(tm, 2, 16));
      tm = fmaxf(tm, __shfl_xor(tm, 4, 16));
      tm = fmaxf(tm, __shfl_xor(tm, 8, 16));
      const float mnew = fmaxf(mrun[r], tm);
      const float esc  = __expf(mrun[r] - mnew);
      mrun[r] = mnew;
      o0[r] *= esc; o1[r] *= esc; lsum[r] *= esc;
      p0[r] = __expf(a - mnew);
      p1[r] = __expf(b - mnew);
      float rs = p0[r] + p1[r];
      rs += __shfl_xor(rs, 1, 16);
      rs += __shfl_xor(rs, 2, 16);
      rs += __shfl_xor(rs, 4, 16);
      rs += __shfl_xor(rs, 8, 16);
      lsum[r] += rs;
    }

    // ---- stage P (C-layout -> LDS), reload as A-fragment; accumulate O += P V ----
    {
      const int hi = (lane & 16) >> 1, n = lane & 15;
#pragma unroll
      for (int r = 0; r < 8; ++r) {
        Ps[wave][r + hi][n]      = (_Float16)p0[r];
        Ps[wave][r + hi][16 + n] = (_Float16)p1[r];
      }
    }
    {
      const int m = lane & 15, kb = (lane & 16) >> 1;
      frag_u pf;
      const _Float16* pp = &Ps[wave][m][kb];
      pf.q[0] = *(const uint4*)pp;                 // k = kb .. kb+7
      pf.q[1] = *(const uint4*)(pp + 16);          // k = kb+16 .. kb+23
      const int d = lane & 15, kk = (lane & 16) >> 1;     // B[k=key][n=d] = V[key][d]
      frag_u bv0, bv1;
      const _Float16* v0p = &Vt[d][kk];
      const _Float16* v1p = &Vt[16 + d][kk];
      bv0.q[0] = *(const uint4*)v0p;  bv0.q[1] = *(const uint4*)(v0p + 16);
      bv1.q[0] = *(const uint4*)v1p;  bv1.q[1] = *(const uint4*)(v1p + 16);
      o0 = __builtin_amdgcn_wmma_f32_16x16x32_f16(false, pf.v, false, bv0.v, (short)0,
                                                  o0, false, false);
      o1 = __builtin_amdgcn_wmma_f32_16x16x32_f16(false, pf.v, false, bv1.v, (short)0,
                                                  o1, false, false);
    }
  }

  // ---- finalize: O /= rowsum, write pixel-major f16 AOt[pixel][head*24+d] ----
  const int n  = lane & 15;
  const int m0 = row0 + ((lane & 16) >> 1);
#pragma unroll
  for (int r = 0; r < 8; ++r) {
    const float inv = 1.0f / lsum[r];
    _Float16* row = AOt + (size_t)(m0 + r) * DIM + head * HD;
    row[n] = (_Float16)(o0[r] * inv);
    if (n < 8) row[16 + n] = (_Float16)(o1[r] * inv);
  }
}

// ---------------------------------------------------------------------------
extern "C" void kernel_launch(void* const* d_in, const int* in_sizes, int n_in,
                              void* d_out, int out_size, void* d_ws, size_t ws_size,
                              hipStream_t stream) {
  const float* x      = (const float*)d_in[0];
  const float* w_qkv  = (const float*)d_in[1];
  const float* w_dw   = (const float*)d_in[2];
  const float* w_proj = (const float*)d_in[3];
  float* out = (float*)d_out;

  char* ws = (char*)d_ws;
  size_t off = 0;
  auto alloc = [&](size_t bytes) -> void* {
    void* p = ws + off;
    off = (off + bytes + 255) & ~(size_t)255;
    return p;
  };
  _Float16* xt  = (_Float16*)alloc((size_t)DIM * NPIX * 2);   // [4096][192]
  _Float16* wqh = (_Float16*)alloc((size_t)3 * DIM * DIM * 2);
  _Float16* wph = (_Float16*)alloc((size_t)DIM * DIM * 2);
  float*    raw = (float*)   alloc((size_t)3 * DIM * NPIX * 4);
  float*    dwc = (float*)   alloc((size_t)3 * DIM * NPIX * 4);
  _Float16* Q   = (_Float16*)alloc((size_t)HEADS * NPIX * HDP * 2);
  _Float16* K   = (_Float16*)alloc((size_t)HEADS * NPIX * HDP * 2);
  _Float16* V   = (_Float16*)alloc((size_t)HEADS * NPIX * HDP * 2);
  _Float16* AOt = (_Float16*)alloc((size_t)DIM * NPIX * 2);   // [4096][192]

  convert_inputs<<<(DIM * NPIX + 255) / 256, 256, 0, stream>>>(x, w_qkv, w_proj,
                                                               xt, wqh, wph);
  // qkv = Wqkv @ x : raw[576][4096]
  wmma_gemm_xt<<<dim3(NPIX / 16, (3 * DIM) / 64), 128, 0, stream>>>(xt, wqh, raw);
  dwconv3x3<<<(3 * DIM * NPIX) / 256, 256, 0, stream>>>(raw, w_dw, dwc);
  normalize_pack<<<(HEADS * NPIX) / 128, 128, 0, stream>>>(dwc, Q, K, V);
  flash_attn<<<dim3(NPIX / 64, HEADS), 128, 0, stream>>>(Q, K, V, AOt);
  // out = Wproj @ attn_out : out[192][4096]
  wmma_gemm_xt<<<dim3(NPIX / 16, DIM / 64), 128, 0, stream>>>(AOt, wph, out);
}
